// MultiHeadAttention_83674552860708
// MI455X (gfx1250) — matrix operand order repaired and checked by reading, not compiled, not run on hardware
//
#include <hip/hip_runtime.h>
#include <hip/hip_bf16.h>

typedef unsigned short u16;
typedef unsigned int   u32;
typedef __attribute__((ext_vector_type(16))) __bf16 v16bf;
typedef __attribute__((ext_vector_type(8)))  float  v8f;
typedef __attribute__((ext_vector_type(4)))  unsigned int v4u;
typedef __attribute__((ext_vector_type(8)))  int  v8i;
typedef __attribute__((ext_vector_type(4)))  int  v4i;

#define EMBED 1024
#define NHEAD 16
#define HD    64
#define SEQN  2048
#define BATCH 4
#define ROWS  (BATCH * SEQN)   // 8192
#define QKSCALE 0.125f

#if __has_builtin(__builtin_amdgcn_tensor_load_to_lds)
#define HAVE_TDM 1
#else
#define HAVE_TDM 0
#endif

// ---------- helpers ----------
__device__ __forceinline__ u16 f2bf(float f) {
    u32 u = __float_as_uint(f);
    u32 r = u + 0x7FFFu + ((u >> 16) & 1u);   // round-to-nearest-even
    return (u16)(r >> 16);
}

union Frag {
    uint4 q[2];
    v16bf v;
};

__device__ __forceinline__ v8f wmma_bf16(v16bf a, v16bf b, v8f c) {
    // D = A(16x32 bf16) * B(32x16 bf16) + C(16x16 f32)
    return __builtin_amdgcn_wmma_f32_16x16x32_bf16(
        false, a, false, b, (short)0, c, false, false);
}

__device__ __forceinline__ Frag lds_frag(const u16* p) {
    Frag f;
    f.q[0] = ((const uint4*)p)[0];
    f.q[1] = ((const uint4*)p)[1];
    return f;
}

// A-fragment read from a row-major [rows][32] bf16 LDS tile.
// lane l: m = l&15, kbase = (l>>4)*16; e0..15 -> k=kbase+0..15
__device__ __forceinline__ Frag a_frag_rowmajor(const u16* tile32, int row, int lg) {
    const u16* p = tile32 + row * 32 + lg * 16;
    Frag f;
    f.q[0] = ((const uint4*)p)[0];
    f.q[1] = ((const uint4*)(p + 8))[0];
    return f;
}

// ============================================================================
// Kernel 1: qkv = x @ W_qkv + b ; RoPE(q,k) ; scatter Q,K [B,H,N,hd] bf16,
//           V transposed [B,H,hd,N] bf16.  Q pre-scaled by SCALE.
// grid = (3072/128, 8192/128), block = 256 (8 waves; wave w owns 16 rows)
// ============================================================================
__global__ __launch_bounds__(256)
void qkv_rope_kernel(const float* __restrict__ x, const float* __restrict__ Wq,
                     const float* __restrict__ bq, const float* __restrict__ rc,
                     const float* __restrict__ rs, u16* __restrict__ Qw,
                     u16* __restrict__ Kw, u16* __restrict__ Vt) {
    __shared__ alignas(16) u16 lA[128][32];     // A tile, row-major bf16
    __shared__ alignas(16) u16 lB[8][32][16];   // B tiles, fragment order

    const int tn   = blockIdx.x * 128;          // output col tile (0..3071)
    const int tm   = blockIdx.y * 128;          // row tile (0..8191)
    const int tid  = threadIdx.x;
    const int wave = tid >> 5;
    const int lane = tid & 31;
    const int l15  = lane & 15;
    const int lg   = lane >> 4;

    v8f acc[8];
#pragma unroll
    for (int j = 0; j < 8; ++j)
#pragma unroll
        for (int i = 0; i < 8; ++i) acc[j][i] = 0.0f;

    const int arow = tid >> 1;                 // A staging: half row per thread
    const int ak0  = (tid & 1) * 16;
    const int bn0  = (tid * 16) & 127;         // B staging
    const int bk   = (tid * 16) >> 7;

    for (int kt = 0; kt < EMBED; kt += 32) {
        __syncthreads();
        // ---- stage A (x fp32 -> bf16, row-major) ----
        {
            const float* src = x + (size_t)(tm + arow) * EMBED + kt + ak0;
            if (kt + 32 < EMBED) __builtin_prefetch(src + 32, 0, 1);
            u16 t[16];
#pragma unroll
            for (int i = 0; i < 16; ++i) t[i] = f2bf(src[i]);
            uint4 lo, hi;
            lo.x = t[0] | (t[1] << 16);  lo.y = t[2] | (t[3] << 16);
            lo.z = t[4] | (t[5] << 16);  lo.w = t[6] | (t[7] << 16);
            hi.x = t[8] | (t[9] << 16);  hi.y = t[10] | (t[11] << 16);
            hi.z = t[12] | (t[13] << 16); hi.w = t[14] | (t[15] << 16);
            *(uint4*)&lA[arow][ak0]     = lo;
            *(uint4*)&lA[arow][ak0 + 8] = hi;
        }
        // ---- stage B (W_qkv fp32 -> bf16, fragment order) ----
        {
            const float* src = Wq + (size_t)(kt + bk) * 3072 + tn + bn0;
            if (kt + 32 < EMBED) __builtin_prefetch(src + 32 * 3072, 0, 1);
#pragma unroll
            for (int i = 0; i < 16; ++i)
                lB[bn0 >> 4][i + 16 * (bk >> 4)][bk & 15] = f2bf(src[i]);
        }
        __syncthreads();
        // ---- compute: preload all fragments, then back-to-back WMMAs ----
        Frag a = a_frag_rowmajor(&lA[0][0], wave * 16 + l15, lg);
        Frag bfr[8];
#pragma unroll
        for (int j = 0; j < 8; ++j) bfr[j] = lds_frag(&lB[j][lane][0]);
        __builtin_amdgcn_sched_barrier(0);   // loads above, WMMAs below
#pragma unroll
        for (int j = 0; j < 8; ++j) acc[j] = wmma_bf16(a.v, bfr[j].v, acc[j]);
        __builtin_amdgcn_sched_barrier(0);
    }

    // ---- epilogue: bias, RoPE, scatter ----
    float bias[8];
#pragma unroll
    for (int j = 0; j < 8; ++j) bias[j] = bq[tn + j * 16 + l15];
#pragma unroll
    for (int j = 0; j < 8; ++j)
#pragma unroll
        for (int r = 0; r < 8; ++r) acc[j][r] += bias[j];

    const int third = tn >> 10;                 // 0=q 1=k 2=v (tile in one third)
    const int h0    = (tn & 1023) >> 6;         // first head in this 128-col tile

    if (third < 2) {
        u16* dst = (third == 0) ? Qw : Kw;
        const float scl = (third == 0) ? QKSCALE : 1.0f;
#pragma unroll
        for (int hh = 0; hh < 2; ++hh) {
#pragma unroll
            for (int jj = 0; jj < 2; ++jj) {
                const int ja = hh * 4 + jj, jb = ja + 2;
                const int d_lo = jj * 16 + l15, d_hi = d_lo + 32;
                const int h = h0 + hh;
#pragma unroll
                for (int r = 0; r < 8; ++r) {
                    const int gm  = tm + wave * 16 + r + 8 * lg;
                    const int b   = gm >> 11;
                    const int seq = gm & (SEQN - 1);
                    const float cl = rc[seq * HD + d_lo], sl = rs[seq * HD + d_lo];
                    const float ch = rc[seq * HD + d_hi], sh = rs[seq * HD + d_hi];
                    const float lo = acc[ja][r], hi = acc[jb][r];
                    const float nlo = (lo * cl - hi * sl) * scl;
                    const float nhi = (hi * ch + lo * sh) * scl;
                    const size_t base = ((size_t)(b * NHEAD + h) * SEQN + seq) * HD;
                    dst[base + d_lo] = f2bf(nlo);
                    dst[base + d_hi] = f2bf(nhi);
                }
            }
        }
    } else {
#pragma unroll
        for (int j = 0; j < 8; ++j) {
            const int col = (tn & 1023) + j * 16 + l15;
            const int h = col >> 6, d = col & 63;
#pragma unroll
            for (int r = 0; r < 8; ++r) {
                const int gm  = tm + wave * 16 + r + 8 * lg;
                const int b   = gm >> 11;
                const int seq = gm & (SEQN - 1);
                Vt[((size_t)(b * NHEAD + h) * HD + d) * SEQN + seq] = f2bf(acc[j][r]);
            }
        }
    }
}

// ============================================================================
// Kernel 2: flash attention per (b,h): O = softmax(Q K^T) V  (Q pre-scaled)
// grid = (16 qtiles, 16 heads, 4 batch), block = 256 (wave owns 16 q rows)
// ============================================================================
__global__ __launch_bounds__(256)
void attn_kernel(const u16* __restrict__ Qw, const u16* __restrict__ Kw,
                 const u16* __restrict__ Vt, u16* __restrict__ attn) {
    __shared__ alignas(16) u16 lP[8][32][16];   // per-wave P scratch

    const int qt = blockIdx.x, h = blockIdx.y, b = blockIdx.z;
    const int wave = threadIdx.x >> 5;
    const int lane = threadIdx.x & 31;
    const int l15 = lane & 15, lg = lane >> 4;

    const size_t bh = (size_t)(b * NHEAD + h);
    const u16* Qp = Qw + bh * SEQN * HD;
    const u16* Kp = Kw + bh * SEQN * HD;
    const u16* Vp = Vt + bh * HD * SEQN;

    const int qrow0 = qt * 128 + wave * 16;

    // Q A-fragments (d = 0..31, 32..63)
    Frag qf[2];
    {
        const u16* pr = Qp + (size_t)(qrow0 + l15) * HD + lg * 16;
#pragma unroll
        for (int fi = 0; fi < 2; ++fi) {
            qf[fi].q[0] = *(const uint4*)(pr + fi * 32);
            qf[fi].q[1] = *(const uint4*)(pr + fi * 32 + 8);
        }
    }

    float mrow[8], lrow[8];
    v8f o[4];
#pragma unroll
    for (int r = 0; r < 8; ++r) { mrow[r] = -INFINITY; lrow[r] = 0.0f; }
#pragma unroll
    for (int j = 0; j < 4; ++j)
#pragma unroll
        for (int i = 0; i < 8; ++i) o[j][i] = 0.0f;

    for (int kb0 = 0; kb0 < SEQN; kb0 += 32) {
        // ---- preload all K fragments (2 key tiles x 2 d-halves) ----
        Frag kf[2][2];
#pragma unroll
        for (int t = 0; t < 2; ++t) {
            const u16* kr = Kp + (size_t)(kb0 + t * 16 + l15) * HD + lg * 16;
            kf[t][0].q[0] = *(const uint4*)(kr);
            kf[t][0].q[1] = *(const uint4*)(kr + 8);
            kf[t][1].q[0] = *(const uint4*)(kr + 32);
            kf[t][1].q[1] = *(const uint4*)(kr + 40);
        }
        // ---- preload all V fragments (4 d-blocks) ----
        Frag vf[4];
#pragma unroll
        for (int j = 0; j < 4; ++j) {
            const u16* vr = Vp + (size_t)(j * 16 + l15) * SEQN + kb0 + lg * 16;
            vf[j].q[0] = *(const uint4*)(vr);
            vf[j].q[1] = *(const uint4*)(vr + 8);
        }
        __builtin_amdgcn_sched_barrier(0);   // all global loads issued first
        // ---- S = Q K^T for 32 keys (two 16x16 tiles) ----
        v8f s[2];
#pragma unroll
        for (int t = 0; t < 2; ++t) {
            v8f c;
#pragma unroll
            for (int i = 0; i < 8; ++i) c[i] = 0.0f;
            c = wmma_bf16(qf[0].v, kf[t][0].v, c);
            c = wmma_bf16(qf[1].v, kf[t][1].v, c);
            s[t] = c;
        }
        // ---- online softmax (row = 16-lane half of the wave) ----
        float alpha[8];
#pragma unroll
        for (int r = 0; r < 8; ++r) {
            float mx = fmaxf(s[0][r], s[1][r]);
#pragma unroll
            for (int off = 1; off < 16; off <<= 1)
                mx = fmaxf(mx, __shfl_xor(mx, off, 32));
            const float mn = fmaxf(mrow[r], mx);
            const float p0 = __expf(s[0][r] - mn);
            const float p1 = __expf(s[1][r] - mn);
            const float a_ = __expf(mrow[r] - mn);
            float rsum = p0 + p1;
#pragma unroll
            for (int off = 1; off < 16; off <<= 1)
                rsum += __shfl_xor(rsum, off, 32);
            lrow[r] = lrow[r] * a_ + rsum;
            mrow[r] = mn;
            alpha[r] = a_;
            s[0][r] = p0;
            s[1][r] = p1;
        }
#pragma unroll
        for (int j = 0; j < 4; ++j)
#pragma unroll
            for (int r = 0; r < 8; ++r) o[j][r] *= alpha[r];

        // ---- P (C-layout f32) -> wave-private LDS in A-fragment order ----
        // Same-wave LDS ops are in-order at the LDS unit: no block barrier.
#pragma unroll
        for (int t = 0; t < 2; ++t)
#pragma unroll
            for (int r = 0; r < 8; ++r) {
                const int m = r + 8 * lg;
                const int k = l15 + 16 * t;
                lP[wave][m + 16 * (k >> 4)][k & 15] =
                    f2bf(s[t][r]);
            }
        __builtin_amdgcn_wave_barrier();
        Frag pf = lds_frag(&lP[wave][lane][0]);
        __builtin_amdgcn_sched_barrier(0);

        // ---- O += P V ----
#pragma unroll
        for (int j = 0; j < 4; ++j) o[j] = wmma_bf16(pf.v, vf[j].v, o[j]);
        __builtin_amdgcn_sched_barrier(0);
    }

    // ---- finalize: O / l, store bf16 into [B*N, EMBED] (col = h*64+d) ----
#pragma unroll
    for (int r = 0; r < 8; ++r) {
        const float inv = 1.0f / lrow[r];
        const int m = r + 8 * lg;
        const size_t row = (size_t)b * SEQN + qt * 128 + wave * 16 + m;
#pragma unroll
        for (int j = 0; j < 4; ++j) {
            const int col = h * HD + j * 16 + l15;
            attn[row * EMBED + col] = f2bf(o[j][r] * inv);
        }
    }
}

// ============================================================================
// Kernel 3: out = attn(bf16) @ W_proj + b_proj  (fp32 out)
// A tile is DMA'd by the Tensor Data Mover (tensor_load_to_lds) when the
// builtin exists; manual copy fallback otherwise.
// grid = (1024/128, 8192/128), block = 256
// ============================================================================
__global__ __launch_bounds__(256)
void proj_kernel(const u16* __restrict__ attn, const float* __restrict__ Wp,
                 const float* __restrict__ bp, float* __restrict__ out) {
    __shared__ alignas(16) u16 lA[128][32];     // A tile, row-major bf16 (8KB)
    __shared__ alignas(16) u16 lB[8][32][16];

    const int tn = blockIdx.x * 128;
    const int tm = blockIdx.y * 128;
    const int tid = threadIdx.x;
    const int wave = tid >> 5, lane = tid & 31;
    const int l15 = lane & 15, lg = lane >> 4;

    v8f acc[8];
#pragma unroll
    for (int j = 0; j < 8; ++j)
#pragma unroll
        for (int i = 0; i < 8; ++i) acc[j][i] = 0.0f;

    const int bn0 = (tid * 16) & 127;
    const int bk  = (tid * 16) >> 7;

#if HAVE_TDM
    // Loop-invariant D# group 1: data_size=2B, tensor 1024x8192 (elem units),
    // tile 32x128, dim0 stride = 1024 elems. (CDNA5 ISA)
    const v8i tdm_g1 = {
        (int)0x00010000u,                                  // data_size=1 (2B)
        (int)((u32)(EMBED & 0xFFFF) << 16),                // tensor_dim0 lo16
        (int)((u32)(ROWS  & 0xFFFF) << 16),                // dim0 hi | dim1 lo16
        (int)((u32)32 << 16),                              // dim1 hi | tile_dim0=32
        128,                                               // tile_dim1=128
        EMBED,                                             // dim0_stride lo32
        0, 0
    };
    const v4i tdm_z4 = {0, 0, 0, 0};
    const u32 lds_base = (u32)(size_t)(&lA[0][0]);         // LDS offset = addr[31:0]
#endif

    for (int kt = 0; kt < EMBED; kt += 32) {
        __syncthreads();
#if HAVE_TDM
        // ---- stage A via Tensor Data Mover (one DMA per tile, wave 0) ----
        if (wave == 0) {
            const unsigned long long ga =
                (unsigned long long)(size_t)(attn + (size_t)tm * EMBED + kt) * 1ull;
            const v4u g0 = {
                1u,                                        // count=1 valid D#
                lds_base,                                  // lds_addr
                (u32)(ga & 0xFFFFFFFFu),                   // global_addr lo
                ((u32)(ga >> 32) & 0x01FFFFFFu) | 0x80000000u  // addr hi | type=2
            };
#if defined(__clang_major__) && (__clang_major__ >= 23)
            __builtin_amdgcn_tensor_load_to_lds(g0, tdm_g1, tdm_z4, tdm_z4,
                                                tdm_g1 * 0, 0);
#else
            __builtin_amdgcn_tensor_load_to_lds(g0, tdm_g1, tdm_z4, tdm_z4, 0);
#endif
            __builtin_amdgcn_s_wait_tensorcnt(0);
        }
#else
        {   // ---- fallback: manual row-major copy (already bf16) ----
            const int arow = tid >> 1;
            const int ak0  = (tid & 1) * 16;
            const u16* src = attn + (size_t)(tm + arow) * EMBED + kt + ak0;
            *(uint4*)&lA[arow][ak0]     = *(const uint4*)src;
            *(uint4*)&lA[arow][ak0 + 8] = *(const uint4*)(src + 8);
        }
#endif
        // ---- stage B (W_proj fp32 -> bf16 fragment order) ----
        {
            const float* src = Wp + (size_t)(kt + bk) * EMBED + tn + bn0;
            if (kt + 32 < EMBED) __builtin_prefetch(src + 32 * EMBED, 0, 1);
#pragma unroll
            for (int i = 0; i < 16; ++i)
                lB[bn0 >> 4][i + 16 * (bk >> 4)][bk & 15] = f2bf(src[i]);
        }
        __syncthreads();
        // ---- compute: preload fragments, back-to-back WMMAs ----
        Frag a = a_frag_rowmajor(&lA[0][0], wave * 16 + l15, lg);
        Frag bfr[8];
#pragma unroll
        for (int j = 0; j < 8; ++j) bfr[j] = lds_frag(&lB[j][lane][0]);
        __builtin_amdgcn_sched_barrier(0);   // loads above, WMMAs below
#pragma unroll
        for (int j = 0; j < 8; ++j) acc[j] = wmma_bf16(a.v, bfr[j].v, acc[j]);
        __builtin_amdgcn_sched_barrier(0);
    }

#pragma unroll
    for (int j = 0; j < 8; ++j) {
        const float bias = bp[tn + j * 16 + l15];
#pragma unroll
        for (int r = 0; r < 8; ++r) {
            const size_t row = (size_t)tm + wave * 16 + r + 8 * lg;
            out[row * EMBED + tn + j * 16 + l15] = acc[j][r] + bias;
        }
    }
}

// ============================================================================
extern "C" void kernel_launch(void* const* d_in, const int* in_sizes, int n_in,
                              void* d_out, int out_size, void* d_ws, size_t ws_size,
                              hipStream_t stream) {
    (void)in_sizes; (void)n_in; (void)out_size; (void)ws_size;
    const float* x  = (const float*)d_in[0];
    const float* rc = (const float*)d_in[1];
    const float* rs = (const float*)d_in[2];
    const float* Wq = (const float*)d_in[3];
    const float* bq = (const float*)d_in[4];
    const float* Wp = (const float*)d_in[5];
    const float* bp = (const float*)d_in[6];
    float* out = (float*)d_out;

    const size_t QK_ELEMS = (size_t)BATCH * NHEAD * SEQN * HD;  // 8.39M bf16
    u16* Qw = (u16*)d_ws;
    u16* Kw = Qw + QK_ELEMS;
    u16* Vt = Kw + QK_ELEMS;
    u16* At = Vt + QK_ELEMS;   // attention output, [B*N, EMBED] bf16

    qkv_rope_kernel<<<dim3(3072 / 128, ROWS / 128), 256, 0, stream>>>(
        x, Wq, bq, rc, rs, Qw, Kw, Vt);
    attn_kernel<<<dim3(SEQN / 128, NHEAD, BATCH), 256, 0, stream>>>(
        Qw, Kw, Vt, At);
    proj_kernel<<<dim3(EMBED / 128, ROWS / 128), 256, 0, stream>>>(
        At, Wp, bp, out);
}
